// CausalSelfAttention_26869315404027
// MI455X (gfx1250) — compile-verified, matrix operand order checked by reading, not run on hardware
//
#include <hip/hip_runtime.h>
#include <hip/hip_bf16.h>

// ---------------------------------------------------------------------------
// CausalSelfAttention forward for gfx1250 (MI455X), bf16 WMMA pipeline.
// B=4, T=2048, C=1024, H=16, D=64.
//   1) f32 -> bf16 conversions
//   2) QKV GEMM (64x64 tile/wave, v_wmma_f32_16x16x32_bf16)
//   3) V transpose -> [B,H,D,T] so P*V B-fragments load as b128
//   4) flash attention (1 wave / 16-row q tile)
//   5) output projection GEMM (f32 out)
// ---------------------------------------------------------------------------

#define ATTN_B 4
#define ATTN_T 2048
#define ATTN_C 1024
#define ATTN_H 16
#define ATTN_D 64
#define ATTN_F3 (3 * ATTN_C)

typedef __attribute__((ext_vector_type(16))) __bf16 v16bf;
typedef __attribute__((ext_vector_type(8)))  __bf16 v8bf;
typedef __attribute__((ext_vector_type(8)))  float  v8f;

// ---- WMMA wrapper ----------------------------------------------------------
__device__ __forceinline__ v8f wmma_bf16(v16bf a, v16bf b, v8f c) {
    // (neg_a, A, neg_b, B, c_mod, C, reuse_a, reuse_b)
    return __builtin_amdgcn_wmma_f32_16x16x32_bf16(false, a, false, b,
                                                   (short)0, c, false, false);
}

// ---- Fragment load ---------------------------------------------------------
// CDNA5 16-bit A/B fragment, per lane: 8 contiguous K-elems at +0 and 8 more
// at +16 (lane halves pre-shifted by caller via khalf). Two b128 loads.
__device__ __forceinline__ v16bf load_frag(const __bf16* __restrict__ p) {
    v8bf lo = *(const v8bf*)(p);
    v8bf hi = *(const v8bf*)(p + 16);
    return __builtin_shufflevector(lo, hi,
        0, 1, 2, 3, 4, 5, 6, 7, 8, 9, 10, 11, 12, 13, 14, 15);
}

// ---- f32 -> bf16 conversion ------------------------------------------------
__global__ void cvt_f32_bf16_kernel(const float* __restrict__ src,
                                    __bf16* __restrict__ dst, int n) {
    int i = blockIdx.x * blockDim.x + threadIdx.x;
    if (i < n) dst[i] = (__bf16)src[i];
}

// ---- V transpose: qkv[b,t,2C + h*64 + d] -> vt[((b*H+h)*D + d)*T + t] -----
__global__ void v_transpose_kernel(const __bf16* __restrict__ qkv,
                                   __bf16* __restrict__ vt) {
    int i = blockIdx.x * blockDim.x + threadIdx.x;           // [B*H*D*T)
    if (i >= ATTN_B * ATTN_H * ATTN_D * ATTN_T) return;
    int t = i & (ATTN_T - 1);
    int r = i >> 11;                 // /T
    int d = r & (ATTN_D - 1);
    r >>= 6;                         // /D
    int h = r & (ATTN_H - 1);
    int b = r >> 4;                  // /H
    vt[i] = qkv[((size_t)b * ATTN_T + t) * ATTN_F3 + 2 * ATTN_C +
                h * ATTN_D + d];
}

// ---- GEMM: Out[M,N] = A[M,K] * W[N,K]^T  (bf16 in, f32 acc) ---------------
// One wave computes a 64x64 tile: 4x4 WMMA accumulators, 16 WMMAs per k-step,
// 16 b128 loads per k-step -> 1 load per WMMA.
template <bool OUT_BF16>
__global__ __launch_bounds__(32)
void gemm_bf16_wmma_kernel(const __bf16* __restrict__ A,
                           const __bf16* __restrict__ W,
                           void* __restrict__ Out,
                           int M, int N, int K) {
    const int lane  = threadIdx.x & 31;
    const int n0    = blockIdx.x * 64;
    const int m0    = blockIdx.y * 64;
    const int rlo   = lane & 15;
    const int khalf = (lane >> 4) << 3;   // 0 or 8

    v8f acc[4][4];
#pragma unroll
    for (int i = 0; i < 4; ++i)
#pragma unroll
        for (int j = 0; j < 4; ++j) acc[i][j] = (v8f){};

    const __bf16* arow = A + (size_t)(m0 + rlo) * K + khalf;
    const __bf16* brow = W + (size_t)(n0 + rlo) * K + khalf;

    for (int k0 = 0; k0 < K; k0 += 32) {
        __builtin_prefetch(arow + k0 + 128, 0, 0);
        __builtin_prefetch(brow + k0 + 128, 0, 0);
        v16bf af[4], bf[4];
#pragma unroll
        for (int i = 0; i < 4; ++i)
            af[i] = load_frag(arow + (size_t)(16 * i) * K + k0);
#pragma unroll
        for (int j = 0; j < 4; ++j)
            bf[j] = load_frag(brow + (size_t)(16 * j) * K + k0);
#pragma unroll
        for (int i = 0; i < 4; ++i)
#pragma unroll
            for (int j = 0; j < 4; ++j)
                acc[i][j] = wmma_bf16(af[i], bf[j], acc[i][j]);
    }

    // C/D layout: lanes 0-15: vgpr j -> M=j; lanes 16-31: vgpr j -> M=8+j.
    const int col   = lane & 15;
    const int rbase = (lane >> 4) << 3;
#pragma unroll
    for (int ti = 0; ti < 4; ++ti)
#pragma unroll
        for (int tj = 0; tj < 4; ++tj)
#pragma unroll
            for (int jj = 0; jj < 8; ++jj) {
                size_t r = (size_t)(m0 + ti * 16 + rbase + jj);
                size_t c = (size_t)(n0 + tj * 16 + col);
                if (OUT_BF16)
                    ((__bf16*)Out)[r * N + c] = (__bf16)acc[ti][tj][jj];
                else
                    ((float*)Out)[r * N + c] = acc[ti][tj][jj];
            }
}

// ---- Flash attention: one wave per (b, h, 16-row q tile) -------------------
// qkv: [B, T, 3C] bf16 (q | k thirds used). vt: [B,H,D,T] bf16 (transposed V).
// out: [B, T, C] bf16.
__global__ __launch_bounds__(32)
void attn_flash_kernel(const __bf16* __restrict__ qkv,
                       const __bf16* __restrict__ vt,
                       __bf16* __restrict__ outbf) {
    const int qt = blockIdx.x;           // q tile (16 rows)
    const int h  = blockIdx.y;
    const int b  = blockIdx.z;
    const int q0 = qt * 16;

    const int lane  = threadIdx.x & 31;
    const int rlo   = lane & 15;
    const int khalf = (lane >> 4) << 3;  // 0 or 8
    const int rbase = (lane >> 4) << 3;  // C-tile row base
    const int col   = lane & 15;

    __shared__ __bf16 pbuf[16 * 32];     // P tile staging (C-layout -> A-frag)

    // Q fragments (16 x 64 -> two k-steps of 32)
    const __bf16* qbase =
        qkv + ((size_t)b * ATTN_T + q0 + rlo) * ATTN_F3 + (size_t)h * ATTN_D;
    v16bf qf0 = load_frag(qbase + khalf);
    v16bf qf1 = load_frag(qbase + 32 + khalf);

    // Vt rows for this lane: vt[((b*H+h)*D + dt*16 + col)*T + ...]
    const __bf16* vtbase =
        vt + ((size_t)(b * ATTN_H + h) * ATTN_D + col) * ATTN_T;

    v8f o0 = {}, o1 = {}, o2 = {}, o3 = {};
    float m_i[8], l_i[8];
#pragma unroll
    for (int j = 0; j < 8; ++j) { m_i[j] = -__builtin_inff(); l_i[j] = 0.f; }

    const float scale = 0.125f;          // 1/sqrt(64)
    const int nblk = (q0 + 15 + 32) / 32;  // 32-key blocks covering keys <= q0+15

    for (int jb = 0; jb < nblk; ++jb) {
        const int j0 = jb * 32;

        // ---- S = Q K^T for 16 rows x 32 keys (2 n-tiles, 2 k-steps) ----
        const __bf16* kb0 = qkv + ((size_t)b * ATTN_T + j0 + rlo) * ATTN_F3 +
                            ATTN_C + (size_t)h * ATTN_D + khalf;
        const __bf16* kb1 = kb0 + (size_t)16 * ATTN_F3;
        v8f s0 = {}, s1 = {};
        {
            v16bf k00 = load_frag(kb0);
            v16bf k10 = load_frag(kb1);
            s0 = wmma_bf16(qf0, k00, s0);
            s1 = wmma_bf16(qf0, k10, s1);
            v16bf k01 = load_frag(kb0 + 32);
            v16bf k11 = load_frag(kb1 + 32);
            s0 = wmma_bf16(qf1, k01, s0);
            s1 = wmma_bf16(qf1, k11, s1);
        }

        // ---- causal mask + online softmax ----
        const int c0 = j0 + col;
        const int c1 = c0 + 16;
        float p0[8], p1[8], corr[8];
#pragma unroll
        for (int jj = 0; jj < 8; ++jj) {
            const int r = q0 + rbase + jj;
            float v0 = (c0 <= r) ? s0[jj] * scale : -__builtin_inff();
            float v1 = (c1 <= r) ? s1[jj] * scale : -__builtin_inff();
            float mx = fmaxf(v0, v1);
#pragma unroll
            for (int off = 8; off >= 1; off >>= 1)
                mx = fmaxf(mx, __shfl_xor(mx, off, 32));  // within 16-lane half
            float nm = fmaxf(m_i[jj], mx);
            corr[jj] = __expf(m_i[jj] - nm);
            p0[jj]   = __expf(v0 - nm);
            p1[jj]   = __expf(v1 - nm);
            float rs = p0[jj] + p1[jj];
#pragma unroll
            for (int off = 8; off >= 1; off >>= 1)
                rs += __shfl_xor(rs, off, 32);
            l_i[jj] = l_i[jj] * corr[jj] + rs;
            m_i[jj] = nm;
        }

        // rescale running output accumulators
#pragma unroll
        for (int jj = 0; jj < 8; ++jj) {
            o0[jj] *= corr[jj]; o1[jj] *= corr[jj];
            o2[jj] *= corr[jj]; o3[jj] *= corr[jj];
        }

        // ---- stage P (C-layout) into LDS, reload as A-fragment ----
#pragma unroll
        for (int jj = 0; jj < 8; ++jj) {
            const int r = rbase + jj;
            pbuf[r * 32 + col]      = (__bf16)p0[jj];
            pbuf[r * 32 + 16 + col] = (__bf16)p1[jj];
        }
        __syncthreads();
        v16bf pf = load_frag(&pbuf[rlo * 32 + khalf]);
        __syncthreads();

        // ---- O += P (16x32) * V (32 keys x 64 dims), 4 n-tiles ----
        // B fragment: lane holds Vt row (dt*16+col), keys j0+khalf+{0..7,16..23}
#pragma unroll
        for (int dt = 0; dt < 4; ++dt) {
            v16bf bv = load_frag(vtbase + (size_t)(dt * 16) * ATTN_T +
                                 j0 + khalf);
            if (dt == 0) o0 = wmma_bf16(pf, bv, o0);
            if (dt == 1) o1 = wmma_bf16(pf, bv, o1);
            if (dt == 2) o2 = wmma_bf16(pf, bv, o2);
            if (dt == 3) o3 = wmma_bf16(pf, bv, o3);
        }
    }

    // ---- normalize and write bf16 output [B,T,C] ----
#pragma unroll
    for (int jj = 0; jj < 8; ++jj) {
        const int t = q0 + rbase + jj;
        const float inv = 1.0f / l_i[jj];
        __bf16* orow = outbf + ((size_t)b * ATTN_T + t) * ATTN_C +
                       (size_t)h * ATTN_D;
        orow[0 * 16 + col] = (__bf16)(o0[jj] * inv);
        orow[1 * 16 + col] = (__bf16)(o1[jj] * inv);
        orow[2 * 16 + col] = (__bf16)(o2[jj] * inv);
        orow[3 * 16 + col] = (__bf16)(o3[jj] * inv);
    }
}

// ---------------------------------------------------------------------------
extern "C" void kernel_launch(void* const* d_in, const int* in_sizes, int n_in,
                              void* d_out, int out_size, void* d_ws, size_t ws_size,
                              hipStream_t stream) {
    (void)in_sizes; (void)n_in; (void)out_size; (void)ws_size;

    const float* x      = (const float*)d_in[0];  // [B,T,C]
    const float* w_qkv  = (const float*)d_in[1];  // [3C,C]
    const float* w_proj = (const float*)d_in[2];  // [C,C]
    float* out = (float*)d_out;                   // [B,T,C]

    const int M  = ATTN_B * ATTN_T;   // 8192
    const int K  = ATTN_C;            // 1024
    const int N1 = ATTN_F3;           // 3072
    const int N2 = ATTN_C;            // 1024

    // workspace layout (bf16 mirrors), 256B aligned
    char* ws = (char*)d_ws;
    size_t off = 0;
    auto take = [&](size_t bytes) {
        char* p = ws + off;
        off += (bytes + 255) & ~(size_t)255;
        return p;
    };
    __bf16* x_bf     = (__bf16*)take((size_t)M * K * 2);        // 16 MB
    __bf16* wqkv_bf  = (__bf16*)take((size_t)N1 * K * 2);       //  6 MB
    __bf16* wproj_bf = (__bf16*)take((size_t)N2 * K * 2);       //  2 MB
    __bf16* qkv_bf   = (__bf16*)take((size_t)M * N1 * 2);       // 48 MB
    __bf16* vt_bf    = (__bf16*)take((size_t)M * ATTN_C * 2);   // 16 MB
    __bf16* attn_bf  = (__bf16*)take((size_t)M * ATTN_C * 2);   // 16 MB

    // 1) f32 -> bf16 conversions
    {
        int n = M * K;
        cvt_f32_bf16_kernel<<<(n + 255) / 256, 256, 0, stream>>>(x, x_bf, n);
        n = N1 * K;
        cvt_f32_bf16_kernel<<<(n + 255) / 256, 256, 0, stream>>>(w_qkv, wqkv_bf, n);
        n = N2 * K;
        cvt_f32_bf16_kernel<<<(n + 255) / 256, 256, 0, stream>>>(w_proj, wproj_bf, n);
    }

    // 2) QKV projection: qkv[M,3C] = x[M,C] * w_qkv[3C,C]^T  (bf16 out)
    {
        dim3 grid(N1 / 64, M / 64);
        gemm_bf16_wmma_kernel<true><<<grid, 32, 0, stream>>>(
            x_bf, wqkv_bf, (void*)qkv_bf, M, N1, K);
    }

    // 3) V transpose -> [B,H,D,T]
    {
        int n = ATTN_B * ATTN_H * ATTN_D * ATTN_T;
        v_transpose_kernel<<<(n + 255) / 256, 256, 0, stream>>>(qkv_bf, vt_bf);
    }

    // 4) Flash attention -> attn_bf [B,T,C]
    {
        dim3 grid(ATTN_T / 16, ATTN_H, ATTN_B);
        attn_flash_kernel<<<grid, 32, 0, stream>>>(qkv_bf, vt_bf, attn_bf);
    }

    // 5) Output projection: out[M,C] = attn[M,C] * w_proj[C,C]^T (f32 out)
    {
        dim3 grid(N2 / 64, M / 64);
        gemm_bf16_wmma_kernel<false><<<grid, 32, 0, stream>>>(
            attn_bf, wproj_bf, (void*)out, M, N2, K);
    }
}
